// Decoder_77833397338146
// MI455X (gfx1250) — matrix operand order repaired and checked by reading, not compiled, not run on hardware
//
#include <hip/hip_runtime.h>
#include <hip/hip_bf16.h>
#include <cstdint>
#include <cstddef>

typedef _Float16 half_t;
typedef __attribute__((ext_vector_type(16))) _Float16 v16h;
typedef __attribute__((ext_vector_type(8)))  float    v8f;
typedef __attribute__((ext_vector_type(4)))  float    f4v;
typedef __attribute__((ext_vector_type(4)))  unsigned int u32x4;
typedef __attribute__((ext_vector_type(8)))  int      i32x8;
typedef __attribute__((ext_vector_type(4)))  int      i32x4;

#define DEV __device__ __forceinline__

#if __has_builtin(__builtin_amdgcn_tensor_load_to_lds) && \
    __has_builtin(__builtin_amdgcn_s_wait_tensorcnt)
#define HAVE_TDM 1
#else
#define HAVE_TDM 0
#endif

constexpr int cV  = 50000;
constexpr int cWD = 256;
constexpr int cE  = 256;
constexpr int cT  = 50;
constexpr int cS  = 200;
constexpr int cB  = 32;
constexpr int cE3 = 3 * cE;     // 768
constexpr int cENC = 2 * cE;    // 512
constexpr int cBT  = cB * cT;   // 1600
constexpr int cBTS = cBT * cS;  // 320000

// ---------------- WMMA fragment helpers (layouts per CDNA5 ISA 7.12.2) -----
// A operand: 16x32 (MxK) f16. Lane L holds row M=L%16; halves 0..7 = K at
// k0+(L/16)*8 .. +7 ; halves 8..15 = same +16.  Row-major K-contiguous source.
DEV v16h load_a_frag(const half_t* base, int ld, int m0, int k0, int lane) {
  const half_t* p = base + (size_t)(m0 + (lane & 15)) * ld + (k0 + ((lane >> 4) << 3));
  v16h f;
  ((f4v*)&f)[0] = *(const f4v*)(p);
  ((f4v*)&f)[1] = *(const f4v*)(p + 16);
  return f;
}
// B operand: 32x16 (KxN) f16 built from W stored [N,K] row-major (so B = W^T).
// Lane L holds column N = n; halves 0..7 = K at k0+(L/16)*8 .. +7 ; halves 8..15 = same +16.
DEV v16h load_b_frag(const half_t* base, int ld, int n, int k0, int lane) {
  const half_t* p = base + (size_t)n * ld + (k0 + ((lane >> 4) << 3));
  v16h f;
  ((f4v*)&f)[0] = *(const f4v*)(p);
  ((f4v*)&f)[1] = *(const f4v*)(p + 16);
  return f;
}

DEV v8f wmma_fma(v16h a, v16h b, v8f c) {
  return __builtin_amdgcn_wmma_f32_16x16x32_f16(false, a, false, b, (short)0, c,
                                                false, false);
}

DEV float sigmoidf_(float x) { return 1.0f / (1.0f + expf(-x)); }

// ---------------- generic GEMM: C[M,N] = A[M,K] * B[N,K]^T (+bias) ---------
// block = 128 threads (4 waves); wave tile 32(M) x 64(N); block tile 128x64.
// No barriers -> OOB waves may simply return. N guarded, M tiles 32-aligned.
__global__ __launch_bounds__(128) void k_gemm_nt(
    const half_t* __restrict__ A, const half_t* __restrict__ B,
    const float* __restrict__ bias, float* __restrict__ C,
    int M, int N, int K) {
  const int lane = threadIdx.x & 31;
  const int wave = threadIdx.x >> 5;
  const int m0 = blockIdx.y * 128 + wave * 32;
  const int n0 = blockIdx.x * 64;
  if (m0 >= M) return;
  const int nl = lane & 15;
  int nr[4];
#pragma unroll
  for (int j = 0; j < 4; ++j) {
    int n = n0 + j * 16 + nl;
    nr[j] = n < N ? n : (N - 1);   // clamp loads; stores guarded below
  }
  v8f acc[2][4] = {};
  for (int k0 = 0; k0 < K; k0 += 32) {
    if (k0 + 32 < K)  // hint next B rows toward near caches (global_prefetch_b8)
      __builtin_prefetch(B + (size_t)nr[0] * K + k0 + 32, 0, 3);
    v16h a0 = load_a_frag(A, K, m0, k0, lane);
    v16h a1 = load_a_frag(A, K, m0 + 16, k0, lane);
#pragma unroll
    for (int j = 0; j < 4; ++j) {
      v16h b = load_b_frag(B, K, nr[j], k0, lane);
      acc[0][j] = wmma_fma(a0, b, acc[0][j]);
      acc[1][j] = wmma_fma(a1, b, acc[1][j]);
    }
  }
#pragma unroll
  for (int i = 0; i < 2; ++i) {
    const int mbase = m0 + i * 16 + ((lane >> 4) << 3);
#pragma unroll
    for (int j = 0; j < 4; ++j) {
      int n = n0 + j * 16 + nl;
      if (n < N) {
        float bv = bias ? bias[n] : 0.0f;
#pragma unroll
        for (int r = 0; r < 8; ++r)
          C[(size_t)(mbase + r) * N + n] = acc[i][j][r] + bv;
      }
    }
  }
}

// ---------------- vocab GEMM with double-buffered TDM-staged B -------------
// Block tile 128(M) x 64(N); K chunked by 128 halves. The Tensor Data Mover
// pulls each 64x128-half B tile into LDS while the block computes on the
// other buffer; s_wait_tensorcnt(0) + barrier publishes it.
DEV void stage_b_chunk(const half_t* __restrict__ B, int N, int K, int n0,
                       int kc, half_t* dst, int tid) {
#if HAVE_TDM
  if ((tid >> 5) == 0) {
    // Tensor DMA Descriptor (CDNA5 ISA 8.3/8.4): 2D tile, data_size=2B,
    // tile 128(K) x 64(rows); tensor dims give OOB zero-fill on ragged N.
    const uint64_t ga = (uint64_t)(uintptr_t)(B + (size_t)n0 * K + kc);
    const uint32_t lds = (uint32_t)(uintptr_t)dst;
    const uint32_t dim0 = (uint32_t)K;           // row length (elements)
    const uint32_t dim1 = (uint32_t)(N - n0);    // remaining rows
    u32x4 g0;
    g0[0] = 1u;                                   // count=1, user mode
    g0[1] = lds;                                  // lds_addr
    g0[2] = (uint32_t)ga;                         // global_addr[31:0]
    g0[3] = (uint32_t)(ga >> 32) | 0x80000000u;   // addr[56:32] | type=2
    i32x8 g1;
    g1[0] = 0x00010000;                           // data_size=1 (2 bytes)
    g1[1] = (int)((dim0 & 0xFFFFu) << 16);        // tensor_dim0 lo16
    g1[2] = (int)(((dim0 >> 16) & 0xFFFFu) | ((dim1 & 0xFFFFu) << 16));
    g1[3] = (int)(((dim1 >> 16) & 0xFFFFu) | (128u << 16));  // tile_dim0=128
    g1[4] = 64;                                   // tile_dim1=64, tile_dim2=0
    g1[5] = (int)dim0;                            // tensor_dim0_stride lo32
    g1[6] = 0;
    g1[7] = 0;
    i32x4 gz = {0, 0, 0, 0};
#if __clang_major__ >= 23
    i32x8 gz8 = {0, 0, 0, 0, 0, 0, 0, 0};
    __builtin_amdgcn_tensor_load_to_lds(g0, g1, gz, gz, gz8, 0);
#else
    __builtin_amdgcn_tensor_load_to_lds(g0, g1, gz, gz, 0);
#endif
  }
#else
  // fallback: cooperative global->LDS copy (256B rows, zero-fill OOB)
  for (int i = tid; i < 64 * 16; i += 128) {
    const int r = i >> 4;
    const int c = (i & 15) << 3;
    const int n = n0 + r;
    f4v val = {};
    if (n < N) val = *(const f4v*)(B + (size_t)n * K + kc + c);
    *(f4v*)(dst + r * 128 + c) = val;
  }
#endif
}

__global__ __launch_bounds__(128) void k_gemm_vocab(
    const half_t* __restrict__ A, const half_t* __restrict__ B,
    const float* __restrict__ bias, float* __restrict__ C,
    int M, int N, int K) {
  __shared__ half_t bs[2][64 * 128];  // double-buffered B chunks (2 x 16 KB)
  const int tid = threadIdx.x;
  const int lane = tid & 31;
  const int wave = tid >> 5;
  const int n0 = blockIdx.x * 64;
  int m0 = blockIdx.y * 128 + wave * 32;
  const bool mval = (m0 < M);
  if (m0 + 32 > M) m0 = M - 32;       // clamp: wave must stay for barriers
  const int nl = lane & 15;

  v8f acc[2][4] = {};
  stage_b_chunk(B, N, K, n0, 0, &bs[0][0], tid);
#if HAVE_TDM
  if (wave == 0) __builtin_amdgcn_s_wait_tensorcnt(0);
#endif
  __syncthreads();
  int buf = 0;
  for (int k0 = 0; k0 < K; k0 += 128) {
    const bool more = (k0 + 128 < K);
    if (more)  // kick off next chunk's DMA while computing on current buffer
      stage_b_chunk(B, N, K, n0, k0 + 128, &bs[buf ^ 1][0], tid);
#pragma unroll
    for (int kk = 0; kk < 128; kk += 32) {
      v16h a0 = load_a_frag(A, K, m0, k0 + kk, lane);
      v16h a1 = load_a_frag(A, K, m0 + 16, k0 + kk, lane);
#pragma unroll
      for (int j = 0; j < 4; ++j) {
        v16h b = load_b_frag(&bs[buf][0], 128, j * 16 + nl, kk, lane);
        acc[0][j] = wmma_fma(a0, b, acc[0][j]);
        acc[1][j] = wmma_fma(a1, b, acc[1][j]);
      }
    }
#if HAVE_TDM
    if (wave == 0 && more) __builtin_amdgcn_s_wait_tensorcnt(0);
#endif
    __syncthreads();
    buf ^= 1;
  }
  if (mval) {
#pragma unroll
    for (int i = 0; i < 2; ++i) {
      const int mbase = m0 + i * 16 + ((lane >> 4) << 3);
#pragma unroll
      for (int j = 0; j < 4; ++j) {
        int n = n0 + j * 16 + nl;
        if (n < N) {
          float bv = bias ? bias[n] : 0.0f;
#pragma unroll
          for (int r = 0; r < 8; ++r)
            C[(size_t)(mbase + r) * N + n] = acc[i][j][r] + bv;
        }
      }
    }
  }
}

// ---------------- GRU recurrent scan (one layer, one persistent block) -----
__global__ __launch_bounds__(256) void k_gru_scan(
    const half_t* __restrict__ Whh,   // [768,256] f16
    const float* __restrict__ bhh,    // [768]
    const float* __restrict__ gi,     // [B*T,768] = x @ Wih^T + bih
    const float* __restrict__ h0,     // [B,256]
    const int*  __restrict__ lengths, // [B]
    float* __restrict__ ghbuf,        // [32,768] scratch
    half_t* __restrict__ xout) {      // [B,T,256] f16
  __shared__ half_t h16[cB][cE];
  __shared__ float  hf[cB][cE];
  const int tid = threadIdx.x, lane = tid & 31, wave = tid >> 5;
  for (int i = tid; i < cB * cE; i += 256) {
    float v = h0[i];
    hf[i >> 8][i & 255]  = v;
    h16[i >> 8][i & 255] = (half_t)v;
  }
  __syncthreads();
  for (int t = 0; t < cT; ++t) {
    for (int tile = wave; tile < 2 * 48; tile += 8) {
      const int m0 = (tile & 1) * 16;
      const int n0 = (tile >> 1) * 16;
      v8f acc = {};
      for (int k0 = 0; k0 < cE; k0 += 32) {
        v16h a = load_a_frag(&h16[0][0], cE, m0, k0, lane);
        v16h b = load_b_frag(Whh, cE, n0 + (lane & 15), k0, lane);
        acc = wmma_fma(a, b, acc);
      }
      const int n = n0 + (lane & 15);
      const int mb = m0 + ((lane >> 4) << 3);
      const float bv = bhh[n];
#pragma unroll
      for (int r = 0; r < 8; ++r)
        ghbuf[(mb + r) * cE3 + n] = acc[r] + bv;
    }
    __syncthreads();
    for (int i = tid; i < cB * cE; i += 256) {
      const int b = i >> 8, e = i & 255;
      const float* gir = gi + (size_t)(b * cT + t) * cE3;
      const float ir = gir[e], iz = gir[e + cE], inn = gir[e + 2 * cE];
      const float hr = ghbuf[b * cE3 + e];
      const float hz = ghbuf[b * cE3 + e + cE];
      const float hn = ghbuf[b * cE3 + e + 2 * cE];
      const float r = sigmoidf_(ir + hr);
      const float z = sigmoidf_(iz + hz);
      const float nn = tanhf(inn + r * hn);
      const float h = hf[b][e];
      const float hnew = (1.0f - z) * nn + z * h;
      const bool mt = t < lengths[b];
      const float hsel = mt ? hnew : h;
      hf[b][e] = hsel;
      h16[b][e] = (half_t)hsel;
      xout[(size_t)(b * cT + t) * cE + e] = (half_t)(mt ? hnew : 0.0f);
    }
    __syncthreads();
  }
}

// ---------------- small utility kernels ------------------------------------
__global__ void k_lengths(const int* __restrict__ tar, int* __restrict__ len) {
  int b = threadIdx.x;
  if (b < cB) {
    int c = 0;
    for (int t = 0; t < cT; ++t) c += (tar[b * cT + t] > 0) ? 1 : 0;
    len[b] = c > 0 ? c : 1;
  }
}

__global__ void k_cast_f16(const float* __restrict__ src, half_t* __restrict__ dst,
                           size_t n) {
  size_t i = (size_t)blockIdx.x * blockDim.x + threadIdx.x;
  if (i < n) dst[i] = (half_t)src[i];
}

__global__ void k_embed(const int* __restrict__ tar, const float* __restrict__ emb,
                        half_t* __restrict__ x16) {
  const int bt = blockIdx.x;        // 0..1599
  const int e = threadIdx.x;        // 0..255
  const int tok = tar[bt];
  x16[(size_t)bt * cE + e] = (half_t)emb[(size_t)tok * cWD + e];
}

__global__ void k_attn_logits(const float* __restrict__ e1, const float* __restrict__ e2,
                              const float* __restrict__ vatt, const int* __restrict__ sou,
                              float* __restrict__ logits) {
  const int gw = (int)((blockIdx.x * (size_t)blockDim.x + threadIdx.x) >> 5);
  const int lane = threadIdx.x & 31;
  if (gw >= cBTS) return;
  const int s = gw % cS;
  const int bt = gw / cS;
  const int b = bt / cT;
  const float* p1 = e1 + (size_t)(b * cS + s) * cE;
  const float* p2 = e2 + (size_t)bt * cE;
  float sum = 0.0f;
  for (int e = lane; e < cE; e += 32) sum += vatt[e] * tanhf(p1[e] + p2[e]);
#pragma unroll
  for (int o = 16; o > 0; o >>= 1) sum += __shfl_xor(sum, o, 32);
  if (lane == 0) {
    if (sou[b * cS + s] == 0) sum = -INFINITY;
    logits[gw] = sum;
  }
}

__global__ void k_softmax_s(float* __restrict__ att) {
  const int row = (int)((blockIdx.x * (size_t)blockDim.x + threadIdx.x) >> 5);
  const int lane = threadIdx.x & 31;
  if (row >= cBT) return;
  float* p = att + (size_t)row * cS;
  float m = -INFINITY;
  for (int s = lane; s < cS; s += 32) m = fmaxf(m, p[s]);
#pragma unroll
  for (int o = 16; o > 0; o >>= 1) m = fmaxf(m, __shfl_xor(m, o, 32));
  float sum = 0.0f;
  for (int s = lane; s < cS; s += 32) {
    float e = expf(p[s] - m);
    p[s] = e;
    sum += e;
  }
#pragma unroll
  for (int o = 16; o > 0; o >>= 1) sum += __shfl_xor(sum, o, 32);
  const float inv = 1.0f / sum;
  for (int s = lane; s < cS; s += 32) p[s] *= inv;
}

__global__ void k_weighted_concat(const float* __restrict__ att,
                                  const float* __restrict__ enc,
                                  const half_t* __restrict__ outp,
                                  half_t* __restrict__ allout) {
  const int d = blockIdx.x * blockDim.x + threadIdx.x;  // 0..767
  const int bt = blockIdx.y;
  if (d >= cE3) return;
  if (d < cENC) {
    const float* a = att + (size_t)bt * cS;
    const float* ep = enc + (size_t)(bt / cT) * cS * cENC + d;
    float s = 0.0f;
    for (int si = 0; si < cS; ++si) s += a[si] * ep[(size_t)si * cENC];
    allout[(size_t)bt * cE3 + d] = (half_t)s;
  } else {
    allout[(size_t)bt * cE3 + d] = outp[(size_t)bt * cE + (d - cENC)];
  }
}

__global__ void k_gate(const half_t* __restrict__ allout, const float* __restrict__ Wg,
                       const float* __restrict__ bg, float* __restrict__ gate) {
  const int bt = (int)((blockIdx.x * (size_t)blockDim.x + threadIdx.x) >> 5);
  const int lane = threadIdx.x & 31;
  if (bt >= cBT) return;
  float sum = 0.0f;
  const half_t* p = allout + (size_t)bt * cE3;
  for (int j = lane; j < cE3; j += 32) sum += (float)p[j] * Wg[j];
#pragma unroll
  for (int o = 16; o > 0; o >>= 1) sum += __shfl_xor(sum, o, 32);
  if (lane == 0) gate[bt] = sigmoidf_(sum + bg[0]);
}

__global__ __launch_bounds__(256) void k_row_stats(const float* __restrict__ X,
                                                   float* __restrict__ rmax,
                                                   float* __restrict__ rsum) {
  __shared__ float sm[256], ss[256];
  const int row = blockIdx.x;
  const int tid = threadIdx.x;
  const float* p = X + (size_t)row * cV;
  float m = -INFINITY, s = 0.0f;
  for (int i = tid; i < cV; i += 256) {
    const float v = p[i];
    if (v > m) { s *= expf(m - v); m = v; }
    s += expf(v - m);
  }
  sm[tid] = m; ss[tid] = s;
  __syncthreads();
  for (int o = 128; o > 0; o >>= 1) {
    if (tid < o) {
      const float m2 = sm[tid + o], s2 = ss[tid + o];
      const float M = fmaxf(sm[tid], m2);
      ss[tid] = ss[tid] * expf(sm[tid] - M) + s2 * expf(m2 - M);
      sm[tid] = M;
    }
    __syncthreads();
  }
  if (tid == 0) { rmax[row] = sm[0]; rsum[row] = ss[0]; }
}

__global__ void k_finalize(float* __restrict__ out, const float* __restrict__ rmax,
                           const float* __restrict__ rsum, const float* __restrict__ gate) {
  const int v = blockIdx.x * blockDim.x + threadIdx.x;
  const int bt = blockIdx.y;
  if (v >= cV) return;
  const size_t i = (size_t)bt * cV + v;
  out[i] = (1.0f - gate[bt]) * expf(out[i] - rmax[bt]) * (1.0f / rsum[bt]);
}

__global__ void k_copy_scatter(const float* __restrict__ att, const int* __restrict__ sou,
                               const float* __restrict__ gate, float* __restrict__ out) {
  const int i = blockIdx.x * blockDim.x + threadIdx.x;
  if (i >= cBTS) return;
  const int s = i % cS;
  const int bt = i / cS;
  const int b = bt / cT;
  const int v = sou[b * cS + s];
  atomicAdd(out + (size_t)bt * cV + v, gate[bt] * att[i]);
}

// ---------------- host orchestration ---------------------------------------
extern "C" void kernel_launch(void* const* d_in, const int* in_sizes, int n_in,
                              void* d_out, int out_size, void* d_ws, size_t ws_size,
                              hipStream_t stream) {
  (void)in_sizes; (void)n_in; (void)out_size; (void)ws_size;
  const int*   sou    = (const int*)d_in[0];
  const int*   tar    = (const int*)d_in[1];
  const float* hidden = (const float*)d_in[2];
  const float* enc    = (const float*)d_in[3];
  const float* emb    = (const float*)d_in[4];
  const float* Wih[3] = {(const float*)d_in[5], (const float*)d_in[9],  (const float*)d_in[13]};
  const float* Whh[3] = {(const float*)d_in[6], (const float*)d_in[10], (const float*)d_in[14]};
  const float* bih[3] = {(const float*)d_in[7], (const float*)d_in[11], (const float*)d_in[15]};
  const float* bhh[3] = {(const float*)d_in[8], (const float*)d_in[12], (const float*)d_in[16]};
  const float* Wenc   = (const float*)d_in[17];
  const float* Wdec   = (const float*)d_in[18];
  const float* vatt   = (const float*)d_in[19];
  const float* Wout   = (const float*)d_in[20];
  const float* bout   = (const float*)d_in[21];
  const float* Wgate  = (const float*)d_in[22];
  const float* bgate  = (const float*)d_in[23];

  float* out = (float*)d_out;                 // [B,T,V]
  float* att = out + (size_t)cBT * cV;        // [B,T,S]

  char* cur = (char*)d_ws;
  auto walloc = [&](size_t bytes) -> void* {
    void* p = (void*)cur;
    cur += (bytes + 255) & ~(size_t)255;
    return p;
  };
  int*    len    = (int*)walloc((size_t)cB * 4);
  half_t* x16[4];
  for (int i = 0; i < 4; ++i) x16[i] = (half_t*)walloc((size_t)cBT * cE * 2);
  float*  gi     = (float*)walloc((size_t)cBT * cE3 * 4);
  float*  ghbuf  = (float*)walloc((size_t)cB * cE3 * 4);
  half_t* Wih16[3];
  half_t* Whh16[3];
  for (int l = 0; l < 3; ++l) {
    Wih16[l] = (half_t*)walloc((size_t)cE3 * cE * 2);
    Whh16[l] = (half_t*)walloc((size_t)cE3 * cE * 2);
  }
  half_t* Wenc16   = (half_t*)walloc((size_t)cE * cENC * 2);
  half_t* Wdec16   = (half_t*)walloc((size_t)cE * cE * 2);
  half_t* Wout16   = (half_t*)walloc((size_t)cV * cE3 * 2);
  half_t* enc16    = (half_t*)walloc((size_t)cB * cS * cENC * 2);
  float*  e1       = (float*)walloc((size_t)cB * cS * cE * 4);
  float*  e2       = (float*)walloc((size_t)cBT * cE * 4);
  half_t* allout16 = (half_t*)walloc((size_t)cBT * cE3 * 2);
  float*  gate     = (float*)walloc((size_t)cBT * 4);
  float*  rmax     = (float*)walloc((size_t)cBT * 4);
  float*  rsum     = (float*)walloc((size_t)cBT * 4);

  auto cast = [&](const float* s, half_t* d, size_t n) {
    k_cast_f16<<<(unsigned)((n + 255) / 256), 256, 0, stream>>>(s, d, n);
  };

  k_lengths<<<1, 32, 0, stream>>>(tar, len);

  for (int l = 0; l < 3; ++l) {
    cast(Wih[l], Wih16[l], (size_t)cE3 * cE);
    cast(Whh[l], Whh16[l], (size_t)cE3 * cE);
  }
  cast(Wenc, Wenc16, (size_t)cE * cENC);
  cast(Wdec, Wdec16, (size_t)cE * cE);
  cast(Wout, Wout16, (size_t)cV * cE3);
  cast(enc,  enc16,  (size_t)cB * cS * cENC);

  k_embed<<<cBT, cE, 0, stream>>>(tar, emb, x16[0]);

  // 3 GRU layers: gi = x @ Wih^T + bih (batched WMMA GEMM), then scan.
  const dim3 gGi((cE3 + 63) / 64, (cBT + 127) / 128);
  for (int l = 0; l < 3; ++l) {
    k_gemm_nt<<<gGi, 128, 0, stream>>>(x16[l], Wih16[l], bih[l], gi, cBT, cE3, cE);
    k_gru_scan<<<1, 256, 0, stream>>>(Whh16[l], bhh[l], gi,
                                      hidden + (size_t)l * cB * cE, len, ghbuf,
                                      x16[l + 1]);
  }

  // attention projections
  k_gemm_nt<<<dim3((cE + 63) / 64, (cB * cS + 127) / 128), 128, 0, stream>>>(
      enc16, Wenc16, nullptr, e1, cB * cS, cE, cENC);
  k_gemm_nt<<<dim3((cE + 63) / 64, (cBT + 127) / 128), 128, 0, stream>>>(
      x16[3], Wdec16, nullptr, e2, cBT, cE, cE);

  // energy -> masked logits -> softmax over S (in place in d_out's atten slot)
  k_attn_logits<<<(cBTS * 32 + 255) / 256, 256, 0, stream>>>(e1, e2, vatt, sou, att);
  k_softmax_s<<<(cBT * 32 + 255) / 256, 256, 0, stream>>>(att);

  // weighted context + concat with decoder output
  k_weighted_concat<<<dim3((cE3 + 255) / 256, cBT), 256, 0, stream>>>(att, enc, x16[3],
                                                                      allout16);
  k_gate<<<(cBT * 32 + 255) / 256, 256, 0, stream>>>(allout16, Wgate, bgate, gate);

  // vocab projection (dominant GEMM): double-buffered TDM-staged B
  k_gemm_vocab<<<dim3((cV + 63) / 64, (cBT + 127) / 128), 128, 0, stream>>>(
      allout16, Wout16, bout, out, cBT, cV, cE3);

  // softmax over V, scale by (1-gate), then add gated copy distribution
  k_row_stats<<<cBT, 256, 0, stream>>>(out, rmax, rsum);
  k_finalize<<<dim3((cV + 255) / 256, cBT), 256, 0, stream>>>(out, rmax, rsum, gate);
  k_copy_scatter<<<(cBTS + 255) / 256, 256, 0, stream>>>(att, sou, gate, out);
}
